// SoftAttention_53077205844200
// MI455X (gfx1250) — compile-verified, matrix operand order checked by reading, not run on hardware
//
#include <hip/hip_runtime.h>

typedef _Float16 half_t;
typedef __attribute__((ext_vector_type(16))) _Float16 v16h;
typedef __attribute__((ext_vector_type(4)))  _Float16 v4h;
typedef __attribute__((ext_vector_type(8)))  float    v8f;

constexpr int B   = 256;
constexpr int L   = 196;
constexpr int ENC = 2048;
constexpr int DEC = 512;
constexpr int ATT = 512;

constexpr int MT = 32;                 // L-rows per block (2 WMMA M-tiles)
constexpr int LT = (L + MT - 1) / MT;  // 7 L-tiles per batch
constexpr int KC = 512;                // K-chunk staged in LDS

// ---------------------------------------------------------------------------
// Kernel 0: pack W_enc [ENC][ATT] (f32) -> Wt [ATT][ENC] (f16), transposed so
// each WMMA B-fragment is a contiguous 32B run of K per lane.
// ---------------------------------------------------------------------------
__global__ __launch_bounds__(256) void pack_wt_kernel(const float* __restrict__ W_enc,
                                                      half_t* __restrict__ Wt) {
    int idx = blockIdx.x * 256 + threadIdx.x;   // over ATT*ENC, k fastest
    int n = idx / ENC;
    int k = idx % ENC;
    Wt[idx] = (half_t)W_enc[(size_t)k * ATT + n];
}

// ---------------------------------------------------------------------------
// Kernel 1: att2[b,n] = b_enc[n] + b_dec[n] + sum_k h[b,k] * W_dec[k,n]
// ---------------------------------------------------------------------------
__global__ __launch_bounds__(256) void att2_kernel(const float* __restrict__ h,
                                                   const float* __restrict__ W_dec,
                                                   const float* __restrict__ b_enc,
                                                   const float* __restrict__ b_dec,
                                                   float* __restrict__ att2) {
    __shared__ float hs[DEC];
    const int b = blockIdx.x, tid = threadIdx.x;
    for (int k = tid; k < DEC; k += 256) hs[k] = h[(size_t)b * DEC + k];
    __syncthreads();
    for (int n = tid; n < ATT; n += 256) {
        float acc = b_enc[n] + b_dec[n];
        for (int k = 0; k < DEC; ++k) acc += hs[k] * W_dec[(size_t)k * ATT + n];
        att2[(size_t)b * ATT + n] = acc;
    }
}

// ---------------------------------------------------------------------------
// Kernel 2: main fused score GEMM.
//   att[b,l] = relu( x[b,l,:] @ W_enc + att2[b,:] ) @ W_full + b_full
// Block = (ltile, b), 256 threads = 8 waves. Each wave owns 64 N-columns.
// x tile staged f32->f16 into LDS per K-chunk; A frags from LDS, B frags from
// packed global Wt. 8 v_wmma_f32_16x16x32_f16 per K-step per wave.
// Next K-chunk is prefetched (global_prefetch_b8) during the WMMA phase.
// ---------------------------------------------------------------------------
__global__ __launch_bounds__(256) void attn_gemm_kernel(const float* __restrict__ x,
                                                        const half_t* __restrict__ Wt,
                                                        const float* __restrict__ att2,
                                                        const float* __restrict__ W_full,
                                                        const float* __restrict__ b_full,
                                                        float* __restrict__ att) {
    __shared__ __align__(32) half_t xs[MT][KC];   // 32 KB
    __shared__ float att_partial[MT];

    const int ltile = blockIdx.x;
    const int b     = blockIdx.y;
    const int tid   = threadIdx.x;
    const int wave  = tid >> 5;
    const int lane  = tid & 31;
    const int lo    = lane & 15;   // A: row within M-tile | B/D: column N
    const int g     = lane >> 4;   // K half-group / D M-half
    const int nb    = wave * 64;   // N base for this wave

    if (tid < MT) att_partial[tid] = 0.0f;

    v8f zero = {};
    v8f acc[2][4];
#pragma unroll
    for (int i = 0; i < 2; ++i)
#pragma unroll
        for (int t = 0; t < 4; ++t) acc[i][t] = zero;

    const float* xbase = x + (size_t)b * L * ENC;

    for (int kc = 0; kc < ENC / KC; ++kc) {
        const int KB = kc * KC;
        // --- stage 32 x 512 f32 -> f16 into LDS (64 elems / thread) ---
#pragma unroll
        for (int it = 0; it < (MT * KC) / (256 * 4); ++it) {   // 16 iters
            int e   = (it * 256 + tid) * 4;
            int row = e / KC;
            int k   = e % KC;
            int l   = ltile * MT + row;
            float4 v = make_float4(0.f, 0.f, 0.f, 0.f);
            if (l < L) v = *(const float4*)(xbase + (size_t)l * ENC + KB + k);
            v4h hv = {(half_t)v.x, (half_t)v.y, (half_t)v.z, (half_t)v.w};
            *(v4h*)&xs[row][k] = hv;
        }
        __syncthreads();

        // --- prefetch next K chunk of x toward caches during WMMA phase ---
        if (kc + 1 < ENC / KC) {
            const int KBn = KB + KC;
#pragma unroll
            for (int it = 0; it < (MT * KC) / (256 * 4); it += 2) {  // every other line
                int e   = (it * 256 + tid) * 4;
                int row = e / KC;
                int k   = e % KC;
                int l   = ltile * MT + row;
                if (l < L)
                    __builtin_prefetch(xbase + (size_t)l * ENC + KBn + k, 0, 1);
            }
        }

        // --- WMMA over this K chunk ---
#pragma unroll 4
        for (int kt = 0; kt < KC / 32; ++kt) {
            const int k0 = kt * 32 + 16 * g;
            v16h a0 = *(const v16h*)&xs[lo][k0];
            v16h a1 = *(const v16h*)&xs[16 + lo][k0];
#pragma unroll
            for (int t = 0; t < 4; ++t) {
                const half_t* bp = Wt + (size_t)(nb + 16 * t + lo) * ENC + KB + kt * 32 + 16 * g;
                v16h bf = *(const v16h*)bp;
                acc[0][t] = __builtin_amdgcn_wmma_f32_16x16x32_f16(
                    false, a0, false, bf, (short)0, acc[0][t], false, false);
                acc[1][t] = __builtin_amdgcn_wmma_f32_16x16x32_f16(
                    false, a1, false, bf, (short)0, acc[1][t], false, false);
            }
        }
        __syncthreads();
    }

    // --- fused epilogue: relu(att1 + att2) . W_full, reduced over N ---
    // D layout: lane lo = N index, VGPR r -> M = 8*g + r
    const float* a2p = att2 + (size_t)b * ATT;
#pragma unroll
    for (int i = 0; i < 2; ++i)
#pragma unroll
        for (int t = 0; t < 4; ++t) {
            int n = nb + 16 * t + lo;
            float a2 = a2p[n];
            float wf = W_full[n];
#pragma unroll
            for (int r = 0; r < 8; ++r) {
                float v = acc[i][t][r] + a2;
                v = (v > 0.0f) ? v * wf : 0.0f;
                atomicAdd(&att_partial[16 * i + 8 * g + r], v);
            }
        }
    __syncthreads();

    if (tid < MT) {
        int l = ltile * MT + tid;
        if (l < L) att[(size_t)b * L + l] = att_partial[tid] + b_full[0];
    }
}

// ---------------------------------------------------------------------------
// Kernel 3: softmax over L per batch + sigmoid gate
// ---------------------------------------------------------------------------
__global__ __launch_bounds__(256) void softmax_gate_kernel(const float* __restrict__ att,
                                                           const float* __restrict__ h,
                                                           const float* __restrict__ W_beta,
                                                           const float* __restrict__ b_beta,
                                                           float* __restrict__ alpha_out,
                                                           float* __restrict__ gate) {
    __shared__ float red[256];
    const int b = blockIdx.x, tid = threadIdx.x;

    float v = (tid < L) ? att[(size_t)b * L + tid] : -1e30f;
    red[tid] = v;
    __syncthreads();
    for (int s = 128; s > 0; s >>= 1) {
        if (tid < s) red[tid] = fmaxf(red[tid], red[tid + s]);
        __syncthreads();
    }
    float m = red[0];
    __syncthreads();

    float e = (tid < L) ? __expf(v - m) : 0.0f;
    red[tid] = e;
    __syncthreads();
    for (int s = 128; s > 0; s >>= 1) {
        if (tid < s) red[tid] += red[tid + s];
        __syncthreads();
    }
    float inv = 1.0f / red[0];
    if (tid < L) alpha_out[(size_t)b * L + tid] = e * inv;
    __syncthreads();

    // gate_beta[b] = sigmoid(h[b] . W_beta + b_beta)
    float p = 0.0f;
    for (int k = tid; k < DEC; k += 256) p += h[(size_t)b * DEC + k] * W_beta[k];
    red[tid] = p;
    __syncthreads();
    for (int s = 128; s > 0; s >>= 1) {
        if (tid < s) red[tid] += red[tid + s];
        __syncthreads();
    }
    if (tid == 0) gate[b] = 1.0f / (1.0f + __expf(-(red[0] + b_beta[0])));
}

// ---------------------------------------------------------------------------
// Kernel 4: out[b,c] = gate[b] * sum_l alpha[b,l] * x[b,l,c]
// Pure HBM pass over x: float4 (b128) loads, 1024 channels per block.
// ---------------------------------------------------------------------------
__global__ __launch_bounds__(256) void zout_kernel(const float* __restrict__ x,
                                                   const float* __restrict__ alpha,
                                                   const float* __restrict__ gate,
                                                   float* __restrict__ out) {
    __shared__ float as[L];
    const int cb = blockIdx.x * 1024, b = blockIdx.y, tid = threadIdx.x;
    for (int l = tid; l < L; l += 256) as[l] = alpha[(size_t)b * L + l];
    __syncthreads();

    const int c = cb + tid * 4;
    float4 acc = make_float4(0.f, 0.f, 0.f, 0.f);
    const float* xp = x + (size_t)b * L * ENC + c;
    for (int l = 0; l < L; ++l) {
        float4 v = *(const float4*)(xp + (size_t)l * ENC);
        float a = as[l];
        acc.x += a * v.x;
        acc.y += a * v.y;
        acc.z += a * v.z;
        acc.w += a * v.w;
    }
    float gb = gate[b];
    float4 r = make_float4(gb * acc.x, gb * acc.y, gb * acc.z, gb * acc.w);
    *(float4*)(out + (size_t)b * ENC + c) = r;
}

// ---------------------------------------------------------------------------
extern "C" void kernel_launch(void* const* d_in, const int* in_sizes, int n_in,
                              void* d_out, int out_size, void* d_ws, size_t ws_size,
                              hipStream_t stream) {
    const float* x      = (const float*)d_in[0];
    const float* h      = (const float*)d_in[1];
    const float* W_enc  = (const float*)d_in[2];
    const float* b_enc  = (const float*)d_in[3];
    const float* W_dec  = (const float*)d_in[4];
    const float* b_dec  = (const float*)d_in[5];
    const float* W_full = (const float*)d_in[6];
    const float* b_full = (const float*)d_in[7];
    const float* W_beta = (const float*)d_in[8];
    const float* b_beta = (const float*)d_in[9];

    char*   ws   = (char*)d_ws;
    half_t* Wt   = (half_t*)ws;                                  // 2 MB
    float*  att2 = (float*)(ws + (size_t)ATT * ENC * sizeof(half_t));
    float*  att  = att2 + (size_t)B * ATT;
    float*  gate = att + (size_t)B * L;

    float* out       = (float*)d_out;
    float* z_out     = out;                    // [B, ENC]
    float* alpha_out = out + (size_t)B * ENC;  // [B, L]

    pack_wt_kernel<<<(ATT * ENC) / 256, 256, 0, stream>>>(W_enc, Wt);
    att2_kernel<<<B, 256, 0, stream>>>(h, W_dec, b_enc, b_dec, att2);
    attn_gemm_kernel<<<dim3(LT, B), 256, 0, stream>>>(x, Wt, att2, W_full, b_full, att);
    softmax_gate_kernel<<<B, 256, 0, stream>>>(att, h, W_beta, b_beta, alpha_out, gate);
    zout_kernel<<<dim3(ENC / 1024, B), 256, 0, stream>>>(x, alpha_out, gate, z_out);
}
